// ProteinFeatures_50251117363664
// MI455X (gfx1250) — matrix operand order repaired and checked by reading, not compiled, not run on hardware
//
#include <hip/hip_runtime.h>
#include <hip/hip_bf16.h>

typedef __attribute__((ext_vector_type(16))) _Float16 v16h;
typedef __attribute__((ext_vector_type(8)))  _Float16 v8h;
typedef __attribute__((ext_vector_type(8)))  float    v8f;

#define BB 4
#define LL 2048
#define KK 30
#define NF 128

// _PAIRS from reference (24 atom-pair indices; slot 0 of D25 is D_neighbors)
__constant__ int PA_c[24] = {0,2,3,4,1,1,1,1,0,0,0,4,4,3,0,2,3,4,2,3,4,2,3,2};
__constant__ int PB_c[24] = {0,2,3,4,0,2,3,4,2,3,4,2,3,2,1,1,1,1,0,0,0,4,4,3};

// ---------------------------------------------------------------------------
// Weight transpose+convert: src[K][128] f32 -> dst[128][Kpad] f16 (zero pad)
// ---------------------------------------------------------------------------
__global__ void convert_wt(const float* __restrict__ src, _Float16* __restrict__ dst,
                           int K, int Kpad) {
  int idx = blockIdx.x * 256 + threadIdx.x;
  int total = NF * Kpad;
  if (idx >= total) return;
  int n = idx / Kpad;
  int k = idx - n * Kpad;
  dst[idx] = (k < K) ? (_Float16)src[k * NF + n] : (_Float16)0.0f;
}

// ---------------------------------------------------------------------------
// Backbone: compute [N, Ca, C, O, Cb] per residue
// ---------------------------------------------------------------------------
__global__ void atoms_kernel(const float* __restrict__ X, float* __restrict__ atoms) {
  int idx = blockIdx.x * 256 + threadIdx.x;
  if (idx >= BB * LL) return;
  const float* x = X + (size_t)idx * 12;
  float N0 = x[0], N1 = x[1], N2 = x[2];
  float A0 = x[3], A1 = x[4], A2 = x[5];
  float C0 = x[6], C1 = x[7], C2 = x[8];
  float O0 = x[9], O1 = x[10], O2 = x[11];
  float b0 = A0 - N0, b1 = A1 - N1, b2 = A2 - N2;
  float c0 = C0 - A0, c1 = C1 - A1, c2 = C2 - A2;
  float a0 = b1 * c2 - b2 * c1;
  float a1 = b2 * c0 - b0 * c2;
  float a2 = b0 * c1 - b1 * c0;
  float Cb0 = -0.58273431f * a0 + 0.56802827f * b0 - 0.54067466f * c0 + A0;
  float Cb1 = -0.58273431f * a1 + 0.56802827f * b1 - 0.54067466f * c1 + A1;
  float Cb2 = -0.58273431f * a2 + 0.56802827f * b2 - 0.54067466f * c2 + A2;
  float* o = atoms + (size_t)idx * 15;
  o[0] = N0; o[1] = N1; o[2] = N2;
  o[3] = A0; o[4] = A1; o[5] = A2;
  o[6] = C0; o[7] = C1; o[8] = C2;
  o[9] = O0; o[10] = O1; o[11] = O2;
  o[12] = Cb0; o[13] = Cb1; o[14] = Cb2;
}

// ---------------------------------------------------------------------------
// Top-K: one block per (b,i). Distance row in LDS, 30 argmin passes with
// index tie-break (matches jax top_k of -D_adjust).
// ---------------------------------------------------------------------------
__global__ void __launch_bounds__(256) topk_kernel(const float* __restrict__ X,
                                                   const float* __restrict__ mask,
                                                   float* __restrict__ Dnbr,
                                                   int* __restrict__ Eidx) {
  __shared__ float dist[LL];
  __shared__ float rv[256];
  __shared__ int   ridx[256];
  const int tid = threadIdx.x;
  const int bi = blockIdx.x;          // = b*L + i
  const int b = bi / LL;
  const float cax = X[((size_t)bi * 4 + 1) * 3 + 0];
  const float cay = X[((size_t)bi * 4 + 1) * 3 + 1];
  const float caz = X[((size_t)bi * 4 + 1) * 3 + 2];
  const float mi = mask[bi];

  float lmax = 0.0f;
  for (int j = tid; j < LL; j += 256) {
    int gj = b * LL + j;
    float dx = cax - X[((size_t)gj * 4 + 1) * 3 + 0];
    float dy = cay - X[((size_t)gj * 4 + 1) * 3 + 1];
    float dz = caz - X[((size_t)gj * 4 + 1) * 3 + 2];
    float m2 = mi * mask[gj];
    float d = m2 * sqrtf(dx * dx + dy * dy + dz * dz + 1e-6f);
    dist[j] = d;
    lmax = fmaxf(lmax, d);
  }
  rv[tid] = lmax;
  __syncthreads();
  for (int s = 128; s > 0; s >>= 1) {
    if (tid < s) rv[tid] = fmaxf(rv[tid], rv[tid + s]);
    __syncthreads();
  }
  const float Dmax = rv[0];
  __syncthreads();
  for (int j = tid; j < LL; j += 256) {
    int gj = b * LL + j;
    float m2 = mi * mask[gj];
    dist[j] = dist[j] + (1.0f - m2) * Dmax;
  }
  __syncthreads();

  for (int kk = 0; kk < KK; ++kk) {
    float bv = 3.0e38f;
    int bj = 0x7fffffff;
    for (int j = tid; j < LL; j += 256) {
      float v = dist[j];
      if (v < bv || (v == bv && j < bj)) { bv = v; bj = j; }
    }
    rv[tid] = bv; ridx[tid] = bj;
    __syncthreads();
    for (int s = 128; s > 0; s >>= 1) {
      if (tid < s) {
        float ov = rv[tid + s]; int oj = ridx[tid + s];
        if (ov < rv[tid] || (ov == rv[tid] && oj < ridx[tid])) { rv[tid] = ov; ridx[tid] = oj; }
      }
      __syncthreads();
    }
    if (tid == 0) {
      Eidx[(size_t)bi * KK + kk] = ridx[0];
      Dnbr[(size_t)bi * KK + kk] = rv[0];
      dist[ridx[0]] = 3.0e38f;
    }
    __syncthreads();
  }
}

// ---------------------------------------------------------------------------
// WMMA helpers (wave32; EXEC all-ones: 32-thread blocks, no divergence here)
// A (LDS, 16 x kpad f16) x Bt (global, 128 x kpad f16) -> tile (LDS 16x128 f32)
//
// Per k-step: issue A ds-loads + all 16 B global-loads as one clause, then run
// the 8 v_wmma back-to-back (one loadcnt wait per k-step instead of per tile).
// ---------------------------------------------------------------------------
__device__ __forceinline__ void wmma_tile(const _Float16* A, int kpad, int ktiles,
                                          const _Float16* __restrict__ Bt,
                                          float* tile, int lane) {
  v8f acc[8] = {};
  const int mrow = lane & 15;
  const int hi = (lane >> 4) & 1;
  const _Float16* arow = A + mrow * kpad + hi * 8;
  const _Float16* brow = Bt + (size_t)mrow * kpad + hi * 8;
  for (int kt = 0; kt < ktiles; ++kt) {
    // A fragment (LDS, 2 x 16B ds_load_b128)
    const v8h* pa = (const v8h*)(arow + kt * 32);
    v8h a0 = pa[0];
    v8h a1 = pa[2];
    // All 8 B fragments for this k-step (16 x global_load_b128, one clause)
    v8h b0[8], b1[8];
#pragma unroll
    for (int nt = 0; nt < 8; ++nt) {
      const _Float16* pb = brow + (size_t)(nt * 16) * kpad + kt * 32;
      b0[nt] = *(const v8h*)pb;
      b1[nt] = *(const v8h*)(pb + 16);
    }
    v16h av;
#pragma unroll
    for (int q = 0; q < 8; ++q) { av[q] = a0[q]; av[q + 8] = a1[q]; }
    // 8 WMMAs back-to-back on the staged fragments
#pragma unroll
    for (int nt = 0; nt < 8; ++nt) {
      v16h bv;
#pragma unroll
      for (int q = 0; q < 8; ++q) { bv[q] = b0[nt][q]; bv[q + 8] = b1[nt][q]; }
      acc[nt] = __builtin_amdgcn_wmma_f32_16x16x32_f16(
          false, av, false, bv, (short)0, acc[nt], false, false);
    }
  }
#pragma unroll
  for (int nt = 0; nt < 8; ++nt)
#pragma unroll
    for (int v = 0; v < 8; ++v)
      tile[(v + hi * 8) * NF + nt * 16 + mrow] = acc[nt][v];
}

// LayerNorm a 16x128 f32 LDS tile per row (+optional bias), write f32 to global
__device__ __forceinline__ void ln_store_f32(const float* tile, const float* bias,
                                             const float* __restrict__ g,
                                             const float* __restrict__ bb,
                                             float* __restrict__ out, int rowBase,
                                             int lane) {
  const int row = lane >> 1;
  const int c0 = (lane & 1) * 64;
  float s = 0.0f, ss = 0.0f;
  for (int c = c0; c < c0 + 64; ++c) {
    float x = tile[row * NF + c] + (bias ? bias[c] : 0.0f);
    s += x; ss += x * x;
  }
  s += __shfl_xor(s, 1);
  ss += __shfl_xor(ss, 1);
  float m = s * (1.0f / NF);
  float var = ss * (1.0f / NF) - m * m;
  float rs = rsqrtf(var + 1e-5f);
  for (int c = c0; c < c0 + 64; ++c) {
    float x = tile[row * NF + c] + (bias ? bias[c] : 0.0f);
    out[(size_t)(rowBase + row) * NF + c] = (x - m) * rs * g[c] + bb[c];
  }
}

// Same but write f16 into an LDS staging buffer (for node-path concat)
__device__ __forceinline__ void ln_store_h16(const float* tile, const float* bias,
                                             const float* __restrict__ g,
                                             const float* __restrict__ bb,
                                             _Float16* dst, int dstStride, int coff,
                                             int lane) {
  const int row = lane >> 1;
  const int c0 = (lane & 1) * 64;
  float s = 0.0f, ss = 0.0f;
  for (int c = c0; c < c0 + 64; ++c) {
    float x = tile[row * NF + c] + bias[c];
    s += x; ss += x * x;
  }
  s += __shfl_xor(s, 1);
  ss += __shfl_xor(ss, 1);
  float m = s * (1.0f / NF);
  float var = ss * (1.0f / NF) - m * m;
  float rs = rsqrtf(var + 1e-5f);
  for (int c = c0; c < c0 + 64; ++c) {
    float x = tile[row * NF + c] + bias[c];
    dst[row * dstStride + coff + c] = (_Float16)((x - m) * rs * g[c] + bb[c]);
  }
}

// ---------------------------------------------------------------------------
// Edge kernel: 1 wave handles 16 edge rows (b,i,k): features -> WMMA -> LN
// ---------------------------------------------------------------------------
__global__ void __launch_bounds__(32) edge_kernel(
    const float* __restrict__ atoms, const float* __restrict__ Dnbr,
    const int* __restrict__ Eidx, const int* __restrict__ resIdx,
    const int* __restrict__ chainLab, const float* __restrict__ posW,
    const float* __restrict__ posB, const _Float16* __restrict__ Bt,
    const float* __restrict__ lng, const float* __restrict__ lnb,
    float* __restrict__ Eout) {
  __shared__ _Float16 featA[16 * 416];
  __shared__ float tile[16 * NF];
  __shared__ float d25[16][25];
  __shared__ float atI[16][15];
  __shared__ float atJ[16][15];
  __shared__ int meta_i[16], meta_j[16], meta_d[16];

  const int lane = threadIdx.x;
  const int r0 = blockIdx.x * 16;

  if (lane < 16) {
    int r = r0 + lane;                        // r = (b*L + i)*K + k
    int gi = r / KK;
    int k = r - gi * KK;
    int b = gi / LL;
    int j = Eidx[(size_t)gi * KK + k];
    int gj = b * LL + j;
    meta_i[lane] = gi;
    meta_j[lane] = gj;
    int off = resIdx[gi] - resIdx[gj];
    int ec = (chainLab[gi] == chainLab[gj]) ? 1 : 0;
    int dcl = off + 32;
    dcl = dcl < 0 ? 0 : (dcl > 64 ? 64 : dcl);
    meta_d[lane] = ec ? dcl : 65;
    d25[lane][0] = Dnbr[(size_t)gi * KK + k];
  }
  __syncthreads();

  for (int e = lane; e < 16 * 15; e += 32) {
    int row = e / 15, c = e - row * 15;
    atI[row][c] = atoms[(size_t)meta_i[row] * 15 + c];
    atJ[row][c] = atoms[(size_t)meta_j[row] * 15 + c];
  }
  __syncthreads();

  for (int e = lane; e < 16 * 24; e += 32) {
    int row = e / 24, t = e - row * 24;
    const float* pi = &atI[row][PA_c[t] * 3];
    const float* pj = &atJ[row][PB_c[t] * 3];
    float dx = pi[0] - pj[0], dy = pi[1] - pj[1], dz = pi[2] - pj[2];
    d25[row][t + 1] = sqrtf(dx * dx + dy * dy + dz * dz + 1e-6f);
  }
  __syncthreads();

  for (int e = lane; e < 16 * 416; e += 32) {
    int row = e / 416, f = e - row * 416;
    float val;
    if (f < 16) {
      val = posW[meta_d[row] * 16 + f] + posB[f];
    } else {
      int p = (f - 16) >> 4;
      int q = (f - 16) & 15;
      float mu = 2.0f + (20.0f / 15.0f) * (float)q;
      float t = (d25[row][p] - mu) * 0.8f;  // /sigma, sigma = 1.25
      val = __expf(-t * t);
    }
    featA[row * 416 + f] = (_Float16)val;
  }
  __syncthreads();

  wmma_tile(featA, 416, 13, Bt, tile, lane);
  __syncthreads();
  ln_store_f32(tile, nullptr, lng, lnb, Eout, r0, lane);
}

// ---------------------------------------------------------------------------
// Node kernel: 1 wave / 16 residues; three staged WMMA GEMMs + LNs
// ---------------------------------------------------------------------------
__global__ void __launch_bounds__(32) node_kernel(
    const int* __restrict__ S, const float* __restrict__ VS,
    const float* __restrict__ aa,
    const _Float16* __restrict__ propWt, const float* __restrict__ propB,
    const float* __restrict__ plg, const float* __restrict__ plb,
    const _Float16* __restrict__ seqWt, const float* __restrict__ seqB,
    const float* __restrict__ slg, const float* __restrict__ slb,
    const _Float16* __restrict__ nodeWt, const float* __restrict__ nodeB,
    const float* __restrict__ nlg, const float* __restrict__ nlb,
    float* __restrict__ Vout) {
  __shared__ _Float16 feat[16 * 1280];
  __shared__ float tile[16 * NF];
  __shared__ _Float16 v12[16 * 256];

  const int lane = threadIdx.x;
  const int r0 = blockIdx.x * 16;

  // Stage 1: V_init = aa_table[S] * (S != 0) ; 566 -> pad 576
  for (int e = lane; e < 16 * 576; e += 32) {
    int row = e / 576, f = e - row * 576;
    float val = 0.0f;
    if (f < 566) {
      int s = S[r0 + row];
      val = (s != 0) ? aa[(size_t)s * 566 + f] : 0.0f;
    }
    feat[row * 576 + f] = (_Float16)val;
  }
  __syncthreads();
  wmma_tile(feat, 576, 18, propWt, tile, lane);
  __syncthreads();
  ln_store_h16(tile, propB, plg, plb, v12, 256, 0, lane);
  __syncthreads();

  // Stage 2: V_S @ seq_W
  for (int e = lane; e < 16 * 1280; e += 32) {
    int row = e / 1280, f = e - row * 1280;
    feat[row * 1280 + f] = (_Float16)VS[(size_t)(r0 + row) * 1280 + f];
  }
  __syncthreads();
  wmma_tile(feat, 1280, 40, seqWt, tile, lane);
  __syncthreads();
  ln_store_h16(tile, seqB, slg, slb, v12, 256, 128, lane);
  __syncthreads();

  // Stage 3: concat(V1, V2) @ node_W
  wmma_tile(v12, 256, 8, nodeWt, tile, lane);
  __syncthreads();
  ln_store_f32(tile, nodeB, nlg, nlb, Vout, r0, lane);
}

// ---------------------------------------------------------------------------
extern "C" void kernel_launch(void* const* d_in, const int* in_sizes, int n_in,
                              void* d_out, int out_size, void* d_ws, size_t ws_size,
                              hipStream_t stream) {
  (void)in_sizes; (void)n_in; (void)out_size; (void)ws_size;
  const float* X     = (const float*)d_in[0];
  const int*   S     = (const int*)d_in[1];
  const float* VS    = (const float*)d_in[2];
  const float* mask  = (const float*)d_in[3];
  const int*   ridx  = (const int*)d_in[4];
  const int*   clab  = (const int*)d_in[5];
  const float* aa    = (const float*)d_in[6];
  const float* posW  = (const float*)d_in[7];
  const float* posB  = (const float*)d_in[8];
  const float* edgeW = (const float*)d_in[9];
  const float* elng  = (const float*)d_in[10];
  const float* elnb  = (const float*)d_in[11];
  const float* propW = (const float*)d_in[12];
  const float* propB = (const float*)d_in[13];
  const float* plng  = (const float*)d_in[14];
  const float* plnb  = (const float*)d_in[15];
  const float* seqW  = (const float*)d_in[16];
  const float* seqB  = (const float*)d_in[17];
  const float* slng  = (const float*)d_in[18];
  const float* slnb  = (const float*)d_in[19];
  const float* nodeW = (const float*)d_in[20];
  const float* nodeB = (const float*)d_in[21];
  const float* nlng  = (const float*)d_in[22];
  const float* nlnb  = (const float*)d_in[23];

  // Output layout: V (B*L*128) | E (B*L*K*128) | E_idx (B*L*K) as int bits
  float* Vout = (float*)d_out;
  float* Eout = Vout + (size_t)BB * LL * NF;
  int* EidxOut = (int*)(Eout + (size_t)BB * LL * KK * NF);

  // Workspace layout (256B-aligned regions), total ~2.03 MB
  char* ws = (char*)d_ws;
  float*     atoms  = (float*)(ws + 0);              // B*L*15 f32 = 491,520 B
  float*     Dnbr   = (float*)(ws + 491520);         // B*L*K f32 = 983,040 B
  _Float16*  edgeWt = (_Float16*)(ws + 1474560);     // 128x416  f16
  _Float16*  propWt = (_Float16*)(ws + 1581056);     // 128x576  f16
  _Float16*  seqWt  = (_Float16*)(ws + 1728512);     // 128x1280 f16
  _Float16*  nodeWt = (_Float16*)(ws + 2056192);     // 128x256  f16

  convert_wt<<<(NF * 416 + 255) / 256, 256, 0, stream>>>(edgeW, edgeWt, 416, 416);
  convert_wt<<<(NF * 576 + 255) / 256, 256, 0, stream>>>(propW, propWt, 566, 576);
  convert_wt<<<(NF * 1280 + 255) / 256, 256, 0, stream>>>(seqW, seqWt, 1280, 1280);
  convert_wt<<<(NF * 256 + 255) / 256, 256, 0, stream>>>(nodeW, nodeWt, 256, 256);

  atoms_kernel<<<(BB * LL + 255) / 256, 256, 0, stream>>>(X, atoms);
  topk_kernel<<<BB * LL, 256, 0, stream>>>(X, mask, Dnbr, EidxOut);

  edge_kernel<<<(BB * LL * KK) / 16, 32, 0, stream>>>(
      atoms, Dnbr, EidxOut, ridx, clab, posW, posB, edgeWt, elng, elnb, Eout);

  node_kernel<<<(BB * LL) / 16, 32, 0, stream>>>(
      S, VS, aa, propWt, propB, plng, plnb, seqWt, seqB, slng, slnb,
      nodeWt, nodeB, nlng, nlnb, Vout);
}
